// Attention_Model_36490042147180
// MI455X (gfx1250) — matrix operand order repaired and checked by reading, not compiled, not run on hardware
//
#include <hip/hip_runtime.h>
#include <hip/hip_bf16.h>

// ---------------- model dimensions ----------------
constexpr int V1 = 12001;
constexpr int Ed = 300;     // embed dim
constexpr int Ep = 320;     // embed dim padded to mult of 32 (WMMA K step)
constexpr int Rr = 1024;    // rnn size
constexpr int Hh = 512;     // att hidden
constexpr int Ff = 2048;    // feat dim
constexpr int Bb = 64;      // batch
constexpr int Tt = 20;      // time steps
constexpr int Ll = 196;     // att locations
constexpr int Vp = 12016;   // V1 padded to mult of 16

// ---------------- WMMA fragment types ----------------
typedef __attribute__((ext_vector_type(16))) __bf16 v16bf;
typedef __attribute__((ext_vector_type(8)))  float  v8f;

union FragAB { v16bf v; uint4 q[2]; };
union FragC  { v8f   v; float f[8]; };

__device__ __forceinline__ unsigned short f32_to_bf16(float x) {
    unsigned int u = __float_as_uint(x);
    unsigned int lsb = (u >> 16) & 1u;
    u += 0x7fffu + lsb;                  // round to nearest even
    return (unsigned short)(u >> 16);
}
__device__ __forceinline__ float bf16_to_f32(unsigned short h) {
    return __uint_as_float(((unsigned int)h) << 16);
}
__device__ __forceinline__ float sigmoidf_(float x) { return 1.0f / (1.0f + __expf(-x)); }

// ---------------- f32 -> bf16 convert with optional zero padding ----------------
__global__ __launch_bounds__(256)
void convert_pad_kernel(const float* __restrict__ src, unsigned short* __restrict__ dst,
                        int src_rows, int src_cols, int dst_rows, int dst_cols) {
    long n = (long)dst_rows * dst_cols;
    long stride = (long)gridDim.x * blockDim.x;
    for (long i = (long)blockIdx.x * blockDim.x + threadIdx.x; i < n; i += stride) {
        int r = (int)(i / dst_cols);
        int c = (int)(i % dst_cols);
        unsigned short o = 0;
        if (r < src_rows && c < src_cols)
            o = f32_to_bf16(src[(size_t)r * src_cols + c]);
        dst[i] = o;
    }
}

// ---------------- zero LSTM state ----------------
__global__ __launch_bounds__(256)
void init_state_kernel(float* __restrict__ h, float* __restrict__ c,
                       unsigned short* __restrict__ h_bf) {
    int i = blockIdx.x * blockDim.x + threadIdx.x;
    if (i < Bb * Rr) { h[i] = 0.f; c[i] = 0.f; h_bf[i] = 0; }
}

// ---------------- bf16 WMMA GEMM:  C(MxN) = A(MxK) * W(NxK)^T ----------------
// One wave computes a (16*MT) x 16 output strip: the B (weight) fragment is
// loaded once per 32-deep K-chunk and reused by MT back-to-back WMMAs.
// A, W row-major bf16 (K contiguous); C f32 row-major. addto: accumulate into C.
template<int MT>
__global__ __launch_bounds__(32)
void gemm_bf16_wmma(const unsigned short* __restrict__ A, int lda,
                    const unsigned short* __restrict__ W, int ldb,
                    float* __restrict__ C, int ldc, int K,
                    const float* __restrict__ bias, int nbias, int addto) {
    const int lane  = threadIdx.x;       // 0..31, wave32
    const int tileN = blockIdx.x * 16;
    const int tileM = blockIdx.y * (16 * MT);
    const int mn    = lane & 15;         // M index for A rows, N index for B rows
    const int khalf = lane >> 4;         // 0 or 1

    // A lane rows: VGPR0-3 hold K = khalf*8 + {0..7}, VGPR4-7 hold +16
    const unsigned short* arow[MT];
#pragma unroll
    for (int mt = 0; mt < MT; ++mt)
        arow[mt] = A + (size_t)(tileM + mt * 16 + mn) * lda + khalf * 8;
    // B lane row: VGPR0-3 hold K = khalf*8 + {0..7}, VGPR4-7 hold +16
    const unsigned short* brow = W + (size_t)(tileN + mn) * ldb + khalf * 8;

    FragC cf[MT];
    if (addto) {
#pragma unroll
        for (int mt = 0; mt < MT; ++mt)
#pragma unroll
            for (int v = 0; v < 8; ++v)
                cf[mt].f[v] = C[(size_t)(tileM + mt * 16 + khalf * 8 + v) * ldc + (tileN + mn)];
    } else {
#pragma unroll
        for (int mt = 0; mt < MT; ++mt)
#pragma unroll
            for (int v = 0; v < 8; ++v) cf[mt].f[v] = 0.f;
    }

    for (int k0 = 0; k0 < K; k0 += 32) {
        FragAB b;
        b.q[0] = *(const uint4*)(brow + k0);
        b.q[1] = *(const uint4*)(brow + k0 + 16);
        __builtin_prefetch(brow + k0 + 32, 0, 3);     // global_prefetch next K-chunk
#pragma unroll
        for (int mt = 0; mt < MT; ++mt) {
            FragAB a;
            a.q[0] = *(const uint4*)(arow[mt] + k0);
            a.q[1] = *(const uint4*)(arow[mt] + k0 + 16);
            cf[mt].v = __builtin_amdgcn_wmma_f32_16x16x32_bf16(
                           false, a.v, false, b.v, (short)0, cf[mt].v, false, false);
        }
    }

#pragma unroll
    for (int mt = 0; mt < MT; ++mt)
#pragma unroll
        for (int v = 0; v < 8; ++v) {
            int row = tileM + mt * 16 + khalf * 8 + v;
            int col = tileN + mn;
            float outv = cf[mt].f[v];
            if (bias) outv += (col < nbias) ? bias[col] : 0.f;
            C[(size_t)row * ldc + col] = outv;
        }
}

// ---------------- gather x_t = embed_w[seq[:,t]] -> bf16, zero-padded to Ep ----------------
__global__ __launch_bounds__(256)
void gather_embed_kernel(const int* __restrict__ seq, int t,
                         const float* __restrict__ embed_w,
                         unsigned short* __restrict__ xt) {
    int i = blockIdx.x * blockDim.x + threadIdx.x;   // Bb*Ep threads
    if (i >= Bb * Ep) return;
    int b = i / Ep, c = i % Ep;
    int tok = seq[b * Tt + t];
    xt[i] = (c < Ed) ? f32_to_bf16(embed_w[(size_t)tok * Ed + c]) : 0;
}

// ---------------- e[b,l] = sum_h tanh(p_att + att_h) * alpha_w + alpha_b ----------------
__global__ __launch_bounds__(256)
void att_e_kernel(const float* __restrict__ p_att, const float* __restrict__ att_h,
                  const float* __restrict__ alpha_w, const float* __restrict__ alpha_b,
                  float* __restrict__ e) {
    int wave = (blockIdx.x * blockDim.x + threadIdx.x) >> 5;   // one wave per (b,l)
    int lane = threadIdx.x & 31;
    if (wave >= Bb * Ll) return;
    int b = wave / Ll;
    const float4* pr = (const float4*)(p_att + (size_t)wave * Hh);
    const float4* ar = (const float4*)(att_h + (size_t)b * Hh);
    const float4* aw = (const float4*)alpha_w;
    float acc = 0.f;
#pragma unroll
    for (int i = lane; i < Hh / 4; i += 32) {       // 4 iterations of float4
        float4 p = pr[i], a = ar[i], w4 = aw[i];
        acc += tanhf(p.x + a.x) * w4.x + tanhf(p.y + a.y) * w4.y +
               tanhf(p.z + a.z) * w4.z + tanhf(p.w + a.w) * w4.w;
    }
#pragma unroll
    for (int off = 16; off; off >>= 1) acc += __shfl_xor(acc, off, 32);
    if (lane == 0) e[wave] = acc + alpha_b[0];
}

// ---------------- softmax over L per batch row ----------------
__global__ __launch_bounds__(256)
void softmax_l_kernel(const float* __restrict__ e, float* __restrict__ w) {
    __shared__ float red[256];
    int tid = threadIdx.x, b = blockIdx.x;
    float x = (tid < Ll) ? e[b * Ll + tid] : -3.4e38f;
    red[tid] = x; __syncthreads();
    for (int s = 128; s; s >>= 1) { if (tid < s) red[tid] = fmaxf(red[tid], red[tid + s]); __syncthreads(); }
    float m = red[0]; __syncthreads();
    float ex = (tid < Ll) ? __expf(x - m) : 0.f;
    red[tid] = ex; __syncthreads();
    for (int s = 128; s; s >>= 1) { if (tid < s) red[tid] += red[tid + s]; __syncthreads(); }
    float inv = 1.f / red[0];
    if (tid < Ll) w[b * Ll + tid] = ex * inv;
}

// ---------------- att_res[b,f] = sum_l w[b,l] * att_feats[b,l,f] ----------------
// 256 threads/block, one block per b; each thread owns 8 consecutive f and
// streams att_feats as 16-byte bf16x8 vectors.
__global__ __launch_bounds__(256)
void att_res_kernel(const float* __restrict__ w,
                    const unsigned short* __restrict__ feats_bf,
                    float* __restrict__ att_res, unsigned short* __restrict__ att_res_bf) {
    int b = blockIdx.x;
    int f8 = threadIdx.x * 8;                      // 256*8 == 2048 == Ff
    const float* wr = w + b * Ll;
    const unsigned short* fb = feats_bf + (size_t)b * Ll * Ff + f8;
    float acc[8];
#pragma unroll
    for (int j = 0; j < 8; ++j) acc[j] = 0.f;
    for (int l = 0; l < Ll; ++l) {
        union { uint4 q; unsigned short u[8]; } ld;
        ld.q = *(const uint4*)(fb + (size_t)l * Ff);
        float wl = wr[l];
#pragma unroll
        for (int j = 0; j < 8; ++j) acc[j] += wl * bf16_to_f32(ld.u[j]);
    }
    union { uint4 q; unsigned short u[8]; } st;
#pragma unroll
    for (int j = 0; j < 8; ++j) {
        att_res[(size_t)b * Ff + f8 + j] = acc[j];
        st.u[j] = f32_to_bf16(acc[j]);
    }
    *(uint4*)(att_res_bf + (size_t)b * Ff + f8) = st.q;
}

// ---------------- LSTM gates / state update ----------------
// sums layout per row (5R): [i | f | o | itr_a | itr_b], itr already has a2c added.
__global__ __launch_bounds__(256)
void gates_kernel(const float* __restrict__ sums, float* __restrict__ c,
                  float* __restrict__ h, unsigned short* __restrict__ h_bf) {
    int i = blockIdx.x * blockDim.x + threadIdx.x;   // Bb*Rr threads
    if (i >= Bb * Rr) return;
    int b = i >> 10, r = i & (Rr - 1);
    const float* srow = sums + (size_t)b * 5 * Rr;
    float ig = sigmoidf_(srow[r]);
    float fg = sigmoidf_(srow[Rr + r]);
    float og = sigmoidf_(srow[2 * Rr + r]);
    float g  = fmaxf(srow[3 * Rr + r], srow[4 * Rr + r]);
    float nc = fg * c[i] + ig * g;
    float nh = og * tanhf(nc);
    c[i] = nc; h[i] = nh; h_bf[i] = f32_to_bf16(nh);
}

// ---------------- log-softmax over V1 -> d_out[b,t,:] ----------------
__global__ __launch_bounds__(256)
void log_softmax_kernel(const float* __restrict__ logits, int t, float* __restrict__ out) {
    __shared__ float red[256];
    int tid = threadIdx.x, b = blockIdx.x;
    const float* row = logits + (size_t)b * Vp;
    float mx = -3.4e38f;
    for (int v = tid; v < V1; v += 256) mx = fmaxf(mx, row[v]);
    red[tid] = mx; __syncthreads();
    for (int s = 128; s; s >>= 1) { if (tid < s) red[tid] = fmaxf(red[tid], red[tid + s]); __syncthreads(); }
    mx = red[0]; __syncthreads();
    float sum = 0.f;
    for (int v = tid; v < V1; v += 256) sum += __expf(row[v] - mx);
    red[tid] = sum; __syncthreads();
    for (int s = 128; s; s >>= 1) { if (tid < s) red[tid] += red[tid + s]; __syncthreads(); }
    float lse = mx + __logf(red[0]);
    float* orow = out + ((size_t)b * Tt + t) * V1;
    for (int v = tid; v < V1; v += 256) orow[v] = row[v] - lse;
}

// ---------------- host orchestration ----------------
static inline size_t align_up(size_t x) { return (x + 255) & ~(size_t)255; }

extern "C" void kernel_launch(void* const* d_in, const int* in_sizes, int n_in,
                              void* d_out, int out_size, void* d_ws, size_t ws_size,
                              hipStream_t stream) {
    (void)in_sizes; (void)n_in; (void)out_size; (void)ws_size;

    const int*   seq       = (const int*)  d_in[0];
    const float* att_feats = (const float*)d_in[1];
    const float* embed_w   = (const float*)d_in[2];
    const float* ctx2att_w = (const float*)d_in[3];
    const float* ctx2att_b = (const float*)d_in[4];
    const float* h2att_w   = (const float*)d_in[5];
    const float* h2att_b   = (const float*)d_in[6];
    const float* alpha_w   = (const float*)d_in[7];
    const float* alpha_b   = (const float*)d_in[8];
    const float* i2h_w     = (const float*)d_in[9];
    const float* i2h_b     = (const float*)d_in[10];
    const float* h2h_w     = (const float*)d_in[11];
    const float* h2h_b     = (const float*)d_in[12];
    const float* a2c_w     = (const float*)d_in[13];
    const float* a2c_b     = (const float*)d_in[14];
    const float* logit_w   = (const float*)d_in[15];
    const float* logit_b   = (const float*)d_in[16];
    float* out = (float*)d_out;

    // workspace bump allocator
    char* ws = (char*)d_ws;
    size_t off = 0;
    auto alloc = [&](size_t bytes) -> void* { void* p = ws + off; off = align_up(off + bytes); return p; };

    unsigned short* feats_bf   = (unsigned short*)alloc((size_t)Bb * Ll * Ff * 2);
    unsigned short* w_ctx2att  = (unsigned short*)alloc((size_t)Hh * Ff * 2);
    unsigned short* w_h2att    = (unsigned short*)alloc((size_t)Hh * Rr * 2);
    unsigned short* w_i2h      = (unsigned short*)alloc((size_t)5 * Rr * Ep * 2);
    unsigned short* w_h2h      = (unsigned short*)alloc((size_t)5 * Rr * Rr * 2);
    unsigned short* w_a2c      = (unsigned short*)alloc((size_t)2 * Rr * Ff * 2);
    unsigned short* w_logit    = (unsigned short*)alloc((size_t)Vp * Rr * 2);
    float*          p_att      = (float*)alloc((size_t)Bb * Ll * Hh * 4);
    float*          h_st       = (float*)alloc((size_t)Bb * Rr * 4);
    float*          c_st       = (float*)alloc((size_t)Bb * Rr * 4);
    unsigned short* h_bf       = (unsigned short*)alloc((size_t)Bb * Rr * 2);
    unsigned short* xt_bf      = (unsigned short*)alloc((size_t)Bb * Ep * 2);
    float*          att_h      = (float*)alloc((size_t)Bb * Hh * 4);
    float*          e_buf      = (float*)alloc((size_t)Bb * Ll * 4);
    float*          w_buf      = (float*)alloc((size_t)Bb * Ll * 4);
    float*          att_res    = (float*)alloc((size_t)Bb * Ff * 4);
    unsigned short* att_res_bf = (unsigned short*)alloc((size_t)Bb * Ff * 2);
    float*          sums       = (float*)alloc((size_t)Bb * 5 * Rr * 4);
    float*          logits     = (float*)alloc((size_t)Bb * Vp * 4);

    const int CB = 256, CG = 2048;
    // one-time bf16 conversions (weights + features), zero-padded where needed
    convert_pad_kernel<<<CG, CB, 0, stream>>>(att_feats, feats_bf, Bb * Ll, Ff, Bb * Ll, Ff);
    convert_pad_kernel<<<CG, CB, 0, stream>>>(ctx2att_w, w_ctx2att, Hh, Ff, Hh, Ff);
    convert_pad_kernel<<<CG, CB, 0, stream>>>(h2att_w,   w_h2att,   Hh, Rr, Hh, Rr);
    convert_pad_kernel<<<CG, CB, 0, stream>>>(i2h_w,     w_i2h,     5 * Rr, Ed, 5 * Rr, Ep);
    convert_pad_kernel<<<CG, CB, 0, stream>>>(h2h_w,     w_h2h,     5 * Rr, Rr, 5 * Rr, Rr);
    convert_pad_kernel<<<CG, CB, 0, stream>>>(a2c_w,     w_a2c,     2 * Rr, Ff, 2 * Rr, Ff);
    convert_pad_kernel<<<CG, CB, 0, stream>>>(logit_w,   w_logit,   V1, Rr, Vp, Rr);
    init_state_kernel<<<(Bb * Rr + CB - 1) / CB, CB, 0, stream>>>(h_st, c_st, h_bf);

    // p_att = att_feats @ ctx2att_w^T + b   (12544 x 512, K=2048), 64-row strips
    gemm_bf16_wmma<4><<<dim3(Hh / 16, (Bb * Ll) / 64), 32, 0, stream>>>(
        feats_bf, Ff, w_ctx2att, Ff, p_att, Hh, Ff, ctx2att_b, Hh, 0);

    for (int t = 0; t < Tt; ++t) {
        gather_embed_kernel<<<(Bb * Ep + CB - 1) / CB, CB, 0, stream>>>(seq, t, embed_w, xt_bf);

        // att_h = h @ h2att^T + b   (64 x 512, K=1024)
        gemm_bf16_wmma<4><<<dim3(Hh / 16, 1), 32, 0, stream>>>(
            h_bf, Rr, w_h2att, Rr, att_h, Hh, Rr, h2att_b, Hh, 0);

        att_e_kernel<<<(Bb * Ll) / 8, CB, 0, stream>>>(p_att, att_h, alpha_w, alpha_b, e_buf);
        softmax_l_kernel<<<Bb, CB, 0, stream>>>(e_buf, w_buf);
        att_res_kernel<<<Bb, CB, 0, stream>>>(w_buf, feats_bf, att_res, att_res_bf);

        // sums = x @ i2h^T + i2h_b   (64 x 5120, K=320)
        gemm_bf16_wmma<4><<<dim3((5 * Rr) / 16, 1), 32, 0, stream>>>(
            xt_bf, Ep, w_i2h, Ep, sums, 5 * Rr, Ep, i2h_b, 5 * Rr, 0);
        // sums += h @ h2h^T + h2h_b  (64 x 5120, K=1024)
        gemm_bf16_wmma<4><<<dim3((5 * Rr) / 16, 1), 32, 0, stream>>>(
            h_bf, Rr, w_h2h, Rr, sums, 5 * Rr, Rr, h2h_b, 5 * Rr, 1);
        // sums[:, 3R:5R] += att_res @ a2c^T + a2c_b   (64 x 2048, K=2048)
        gemm_bf16_wmma<4><<<dim3((2 * Rr) / 16, 1), 32, 0, stream>>>(
            att_res_bf, Ff, w_a2c, Ff, sums + 3 * Rr, 5 * Rr, Ff, a2c_b, 2 * Rr, 1);

        gates_kernel<<<(Bb * Rr + CB - 1) / CB, CB, 0, stream>>>(sums, c_st, h_st, h_bf);

        // logits = nh @ logit^T + logit_b   (64 x 12016pad, K=1024)
        gemm_bf16_wmma<4><<<dim3(Vp / 16, 1), 32, 0, stream>>>(
            h_bf, Rr, w_logit, Rr, logits, Vp, Rr, logit_b, V1, 0);

        log_softmax_kernel<<<Bb, CB, 0, stream>>>(logits, t, out);
    }
}
